// WeightedCausalMultiheadAttention_43920335569590
// MI455X (gfx1250) — matrix operand order repaired and checked by reading, not compiled, not run on hardware
//
#include <hip/hip_runtime.h>

// ---------------------------------------------------------------------------
// CDNA5 (gfx1250) bf16 WMMA helpers.  Layouts per CDNA5 ISA:
//   A (16x32, 16-bit): lane L holds row M=L%16; lanes<16: K={0..7,16..23},
//                      lanes>=16: K={8..15,24..31}; pairs per VGPR.
//   B (32x16, 16-bit): lane L holds col N=L%16; lanes<16: K={0..7,16..23},
//                      lanes>=16: K={8..15,24..31} (same as A).
//   C/D (16x16 f32):   VGPR r: lanes<16 -> row r, lanes>=16 -> row r+8; col=lane%16.
// ---------------------------------------------------------------------------
typedef __attribute__((ext_vector_type(16))) __bf16 bf16x16;
typedef __attribute__((ext_vector_type(8)))  __bf16 bf16x8;
typedef __attribute__((ext_vector_type(8)))  float  f32x8;

#define DEV __device__ __forceinline__

DEV f32x8 zero8() {
  f32x8 z;
#pragma unroll
  for (int i = 0; i < 8; ++i) z[i] = 0.0f;
  return z;
}

// p points at (row, k=0) of a K-major row; 16B-aligned.
DEV bf16x16 frag_a(const __bf16* p, int lane) {
  const int kb = (lane & 16) ? 8 : 0;
  bf16x8 lo = *(const bf16x8*)(p + kb);        // K = kb .. kb+7
  bf16x8 hi = *(const bf16x8*)(p + kb + 16);   // K = kb+16 .. kb+23
  bf16x16 r;
#pragma unroll
  for (int i = 0; i < 8; ++i) { r[i] = lo[i]; r[i + 8] = hi[i]; }
  return r;
}

// p points at (colN-row, k=0) of a K-major row of B^T storage; 16B-aligned.
DEV bf16x16 frag_b(const __bf16* p, int lane) {
  const int kb = (lane & 16) ? 8 : 0;
  bf16x8 lo = *(const bf16x8*)(p + kb);        // K = kb .. kb+7
  bf16x8 hi = *(const bf16x8*)(p + kb + 16);   // K = kb+16 .. kb+23
  bf16x16 r;
#pragma unroll
  for (int i = 0; i < 8; ++i) { r[i] = lo[i]; r[i + 8] = hi[i]; }
  return r;
}

DEV f32x8 wmma_bf16(bf16x16 a, bf16x16 b, f32x8 c) {
  return __builtin_amdgcn_wmma_f32_16x16x32_bf16(
      /*neg_a=*/false, a, /*neg_b=*/false, b,
      /*c_mod=*/(short)0, c, /*reuse_a=*/false, /*reuse_b=*/false);
}

// Async global -> LDS copy of 16 bytes per lane (GLOBAL_LOAD_ASYNC_TO_LDS_B128,
// GV mode, tracked by ASYNCcnt).  LDS address = low 32 bits of generic pointer
DEV void async_copy_b128(const __bf16* gsrc, __bf16* ldst) {
  unsigned int l = (unsigned int)(unsigned long long)ldst;
  asm volatile("global_load_async_to_lds_b128 %0, %1, off"
               :: "v"(l), "v"(gsrc)
               : "memory");
}

DEV void async_wait_all() {
  asm volatile("s_wait_asynccnt 0" ::: "memory");
}

// ---------------------------------------------------------------------------
// GEMM:  out = A @ W^T + bias.   A: M x K (AT dtype), W: N x K fp32 (torch
// Linear weight), bias: N fp32.  Block tile 128x128, k-tile 32, 8 waves.
// MODE 0: fp32 row-major out.
// MODE 1: bf16 out head-split:      (b,t,h*128+d) -> [(b*16+h)][t][d]
// MODE 2: bf16 out head-split + T:  (b,t,h*128+d) -> [(b*16+h)][d][t]
//         (T=2048, HD=128, H=16 hardcoded).
// ---------------------------------------------------------------------------
template <typename AT, int MODE>
__global__ __launch_bounds__(256) void gemm_xwT_bias(
    const AT* __restrict__ A, const float* __restrict__ W,
    const float* __restrict__ bias, void* __restrict__ outv,
    int M, int N, int K) {
  constexpr int BM = 128, BN = 128, BK = 32, PAD = 8;
  __shared__ __attribute__((aligned(16))) __bf16 As[BM][BK + PAD];
  __shared__ __attribute__((aligned(16))) __bf16 Bs[BN][BK + PAD];

  const int tid  = threadIdx.x;
  const int lane = tid & 31;
  const int w    = tid >> 5;            // 8 waves
  const int waveM = (w & 3) * 32;       // 4 waves along M (32 rows each)
  const int waveN = (w >> 2) * 64;      // 2 waves along N (64 cols each)
  const int cn   = lane & 15;
  const int hl   = (lane >> 4) & 1;

  const int mBase = blockIdx.y * BM;
  const int nBase = blockIdx.x * BN;

  f32x8 acc[2][4];
#pragma unroll
  for (int mi = 0; mi < 2; ++mi)
#pragma unroll
    for (int ni = 0; ni < 4; ++ni) acc[mi][ni] = zero8();

  const int srow = tid >> 1;            // 128 rows, 2 threads/row
  const int scol = (tid & 1) * 16;      // each thread 16 contiguous k

  for (int kb = 0; kb < K; kb += BK) {
    const AT*    aSrc = A + (size_t)(mBase + srow) * K + kb + scol;
    const float* bSrc = W + (size_t)(nBase + srow) * K + kb + scol;
    __builtin_prefetch(aSrc + BK, 0, 0);   // global_prefetch_b8 (next k-tile)
    __builtin_prefetch(bSrc + BK, 0, 0);
#pragma unroll
    for (int e = 0; e < 16; ++e) As[srow][scol + e] = (__bf16)(float)aSrc[e];
#pragma unroll
    for (int e = 0; e < 16; ++e) Bs[srow][scol + e] = (__bf16)bSrc[e];
    __syncthreads();

    bf16x16 af[2];
#pragma unroll
    for (int mi = 0; mi < 2; ++mi)
      af[mi] = frag_a(&As[waveM + mi * 16 + cn][0], lane);
#pragma unroll
    for (int ni = 0; ni < 4; ++ni) {
      bf16x16 bf = frag_b(&Bs[waveN + ni * 16 + cn][0], lane);
#pragma unroll
      for (int mi = 0; mi < 2; ++mi)
        acc[mi][ni] = wmma_bf16(af[mi], bf, acc[mi][ni]);
    }
    __syncthreads();
  }

#pragma unroll
  for (int mi = 0; mi < 2; ++mi) {
#pragma unroll
    for (int ni = 0; ni < 4; ++ni) {
#pragma unroll
      for (int r = 0; r < 8; ++r) {
        const int m = mBase + waveM + mi * 16 + r + 8 * hl;
        const int n = nBase + waveN + ni * 16 + cn;
        const float v = acc[mi][ni][r] + bias[n];
        if constexpr (MODE == 0) {
          ((float*)outv)[(size_t)m * N + n] = v;
        } else {
          const int b = m >> 11, t = m & 2047;   // M = B*T, T=2048
          const int h = n >> 7,  d = n & 127;    // N = H*HD, HD=128
          if constexpr (MODE == 1) {
            ((__bf16*)outv)[(((size_t)(b * 16 + h)) * 2048 + t) * 128 + d] =
                (__bf16)v;
          } else {  // MODE 2: per-head transposed [head][d][t]
            ((__bf16*)outv)[(((size_t)(b * 16 + h)) * 128 + d) * 2048 + t] =
                (__bf16)v;
          }
        }
      }
    }
  }
}

// ---------------------------------------------------------------------------
// Flash attention with causal mask + power-law decay (alpha=1).
// Reference swaps roles: S[i,j] = K_i . Q_j, so:
//   Qside = K-projection [head][t][d], Kside = Q-projection [head][t][d],
//   Vt    = V-projection stored transposed [head][d][t].
// One block per (head, 128-row block); 8 waves, 16 query rows per wave.
// K/V tiles staged with async global->LDS b128 copies (ASYNCcnt).
// ---------------------------------------------------------------------------
__global__ __launch_bounds__(256) void attn_decay_kernel(
    const __bf16* __restrict__ Qside, const __bf16* __restrict__ Kside,
    const __bf16* __restrict__ Vt, __bf16* __restrict__ Out) {
  constexpr int T = 2048, HD = 128, JB = 64;
  constexpr float SCALE = 0.08838834764831845f;  // 128^-0.5

  __shared__ __attribute__((aligned(16))) __bf16 Ksm[JB][HD + 8];      // keys (j x d)
  __shared__ __attribute__((aligned(16))) __bf16 Vtm[HD][JB + 8];      // V^T (d x j)
  __shared__ __attribute__((aligned(16))) __bf16 Psm[8][16][JB + 8];   // per-wave P

  const int tid  = threadIdx.x;
  const int lane = tid & 31;
  const int w    = tid >> 5;
  const int cn   = lane & 15;
  const int hl   = (lane >> 4) & 1;

  const int ib = blockIdx.x;                 // 128-row block
  const int hh = blockIdx.y;                 // flattened head b*16+h
  const int rowBase = ib * 128 + w * 16;
  const size_t headOff = (size_t)hh * T * HD;   // same byte extent for Vt

  // Query fragments for this wave's 16 rows, all of d=128 (4 k-steps).
  bf16x16 af[4];
#pragma unroll
  for (int ks = 0; ks < 4; ++ks)
    af[ks] = frag_a(Qside + headOff + (size_t)(rowBase + cn) * HD + ks * 32, lane);

  f32x8 Oacc[8];
#pragma unroll
  for (int td = 0; td < 8; ++td) Oacc[td] = zero8();
  float mprev[8], lsum[8];
#pragma unroll
  for (int r = 0; r < 8; ++r) { mprev[r] = -__builtin_inff(); lsum[r] = 0.0f; }

  const int jbMax = 2 * ib + 1;              // causal: j <= max row in block
  for (int jb = 0; jb <= jbMax; ++jb) {
    const int j0 = jb * JB;
    __syncthreads();
    // Stage keys (j x d): 64 rows x 128 bf16 = 1024 b128 chunks.
    for (int c = tid; c < (JB * HD) / 8; c += 256) {
      const int row = c >> 4, c8 = (c & 15) * 8;
      async_copy_b128(Kside + headOff + (size_t)(j0 + row) * HD + c8,
                      &Ksm[row][c8]);
    }
    // Stage V^T (d x j): 128 rows x 64 bf16 = 1024 b128 chunks (no transpose
    // needed -- the V projection GEMM already stored [head][d][t]).
    for (int c = tid; c < (HD * JB) / 8; c += 256) {
      const int d = c >> 3, c8 = (c & 7) * 8;
      async_copy_b128(Vt + headOff + (size_t)d * T + j0 + c8, &Vtm[d][c8]);
    }
    async_wait_all();
    __syncthreads();

    // S strip: 16 x 64 per wave.
    f32x8 s[4];
#pragma unroll
    for (int tn = 0; tn < 4; ++tn) {
      f32x8 a = zero8();
#pragma unroll
      for (int ks = 0; ks < 4; ++ks)
        a = wmma_bf16(af[ks], frag_b(&Ksm[tn * 16 + cn][ks * 32], lane), a);
      s[tn] = a;
    }

    // Scale + causal mask + decay -(i-j); per-row max.
    float mcur[8];
#pragma unroll
    for (int r = 0; r < 8; ++r) mcur[r] = -__builtin_inff();
#pragma unroll
    for (int tn = 0; tn < 4; ++tn) {
#pragma unroll
      for (int r = 0; r < 8; ++r) {
        const int i = rowBase + r + 8 * hl;
        const int j = j0 + tn * 16 + cn;
        const int delta = i - j;
        const float v = (delta < 0) ? -__builtin_inff()
                                    : s[tn][r] * SCALE - (float)delta;
        s[tn][r] = v;
        mcur[r] = fmaxf(mcur[r], v);
      }
    }
#pragma unroll
    for (int r = 0; r < 8; ++r)
#pragma unroll
      for (int off = 1; off < 16; off <<= 1)
        mcur[r] = fmaxf(mcur[r], __shfl_xor(mcur[r], off, 32));

    float corr[8], ladd[8];
#pragma unroll
    for (int r = 0; r < 8; ++r) {
      const float mn = fmaxf(mprev[r], mcur[r]);
      corr[r] = __expf(mprev[r] - mn);
      mprev[r] = mn;
      ladd[r] = 0.0f;
    }
    // P = exp(S - m): accumulate row sums and spill bf16 strip to LDS.
#pragma unroll
    for (int tn = 0; tn < 4; ++tn) {
#pragma unroll
      for (int r = 0; r < 8; ++r) {
        const float p = __expf(s[tn][r] - mprev[r]);
        ladd[r] += p;
        Psm[w][r + 8 * hl][tn * 16 + cn] = (__bf16)p;
      }
    }
#pragma unroll
    for (int r = 0; r < 8; ++r) {
#pragma unroll
      for (int off = 1; off < 16; off <<= 1)
        ladd[r] += __shfl_xor(ladd[r], off, 32);
      lsum[r] = lsum[r] * corr[r] + ladd[r];
    }
#pragma unroll
    for (int td = 0; td < 8; ++td)
#pragma unroll
      for (int r = 0; r < 8; ++r) Oacc[td][r] *= corr[r];

    // Same-wave DS store -> DS load ordering before P is re-read as A frags.
    asm volatile("s_wait_dscnt 0" ::: "memory");

    // O += P @ V   (JB=64 => 2 k-steps of 32)
#pragma unroll
    for (int jks = 0; jks < 2; ++jks) {
      bf16x16 ap = frag_a(&Psm[w][cn][jks * 32], lane);
#pragma unroll
      for (int td = 0; td < 8; ++td)
        Oacc[td] =
            wmma_bf16(ap, frag_b(&Vtm[td * 16 + cn][jks * 32], lane), Oacc[td]);
    }
  }

  // Normalize and write (B,T,D) bf16 with head interleave.
  const int b = hh >> 4, h = hh & 15;
#pragma unroll
  for (int td = 0; td < 8; ++td) {
#pragma unroll
    for (int r = 0; r < 8; ++r) {
      const int i = rowBase + r + 8 * hl;
      const int d = td * 16 + cn;
      const float v = Oacc[td][r] / lsum[r];
      Out[((size_t)(b * T + i)) * 2048 + h * HD + d] = (__bf16)v;
    }
  }
}

// ---------------------------------------------------------------------------
// Host launcher.
// ---------------------------------------------------------------------------
extern "C" void kernel_launch(void* const* d_in, const int* in_sizes, int n_in,
                              void* d_out, int out_size, void* d_ws,
                              size_t ws_size, hipStream_t stream) {
  (void)in_sizes; (void)n_in; (void)out_size; (void)ws_size;
  const float* x   = (const float*)d_in[0];
  const float* q_w = (const float*)d_in[1];
  const float* q_b = (const float*)d_in[2];
  const float* k_w = (const float*)d_in[3];
  const float* k_b = (const float*)d_in[4];
  const float* v_w = (const float*)d_in[5];
  const float* v_b = (const float*)d_in[6];
  const float* o_w = (const float*)d_in[7];
  const float* o_b = (const float*)d_in[8];

  const int M = 4096, N = 2048, K = 2048;      // M = B*T
  const size_t projBytes = (size_t)32 * 2048 * 128 * sizeof(__bf16);  // 16 MB
  char* ws = (char*)d_ws;
  __bf16* Pq   = (__bf16*)(ws + 0 * projBytes);   // Q projection [head][t][d]
  __bf16* Pk   = (__bf16*)(ws + 1 * projBytes);   // K projection [head][t][d]
  __bf16* Pvt  = (__bf16*)(ws + 2 * projBytes);   // V projection [head][d][t]
  __bf16* Attn = (__bf16*)(ws + 3 * projBytes);   // (B,T,D) bf16

  dim3 gGemm(N / 128, M / 128);   // 16 x 32
  dim3 block(256);

  gemm_xwT_bias<float, 1><<<gGemm, block, 0, stream>>>(x, q_w, q_b, Pq, M, N, K);
  gemm_xwT_bias<float, 1><<<gGemm, block, 0, stream>>>(x, k_w, k_b, Pk, M, N, K);
  gemm_xwT_bias<float, 2><<<gGemm, block, 0, stream>>>(x, v_w, v_b, Pvt, M, N, K);

  // S[i,j] = K_i . Q_j  => query-side = K projection, key-side = Q projection.
  dim3 gAttn(16, 32);             // (row blocks of 128, B*H heads)
  attn_decay_kernel<<<gAttn, block, 0, stream>>>(Pk, Pq, Pvt, Attn);

  gemm_xwT_bias<__bf16, 0><<<gGemm, block, 0, stream>>>(Attn, o_w, o_b,
                                                        (float*)d_out, M, N, K);
}
